// TransformerNet_79216376808036
// MI455X (gfx1250) — compile-verified, matrix operand order checked by reading, not run on hardware
//
#include <hip/hip_runtime.h>
#include <math.h>

typedef __attribute__((ext_vector_type(16))) _Float16 v16h;
typedef __attribute__((ext_vector_type(8)))  _Float16 v8h;
typedef __attribute__((ext_vector_type(8)))  float    v8f;

#define NNODES  10000
#define NEDGES  160000
#define NHEADS  4
#define NGRAPHS 64

// ---------------------------------------------------------------- utilities

__global__ void cvt_f32_f16_kernel(const float* __restrict__ s, _Float16* __restrict__ d, int n) {
    int i = blockIdx.x * blockDim.x + threadIdx.x;
    if (i < n) d[i] = (_Float16)s[i];
}

// w [K, Ncols] f32 row-major  ->  wt [Ncols, K] f16 (i.e. column-major B)
__global__ void cvt_transpose_f32_f16_kernel(const float* __restrict__ w,
                                             _Float16* __restrict__ wt,
                                             int K, int Ncols) {
    int t = blockIdx.x * blockDim.x + threadIdx.x;   // output-linear: coalesced stores
    if (t >= K * Ncols) return;
    int col = t / K;
    int k   = t - col * K;
    wt[t] = (_Float16)w[(size_t)k * Ncols + col];
}

__global__ void fill_kernel(float* __restrict__ p, float v, int n) {
    int i = blockIdx.x * blockDim.x + threadIdx.x;
    if (i < n) p[i] = v;
}

__device__ __forceinline__ void atomicMaxF(float* addr, float val) {
    // monotonic int mapping: works with -inf init, mixed signs
    if (val >= 0.0f) atomicMax((int*)addr, __float_as_int(val));
    else             atomicMin((unsigned int*)addr, (unsigned int)__float_as_int(val));
}

// ---------------------------------------------------------------- WMMA GEMM
// C[M,Ncols] = A_f16[M,K] @ B, with B supplied TRANSPOSED: Bt[Ncols,K] f16.
// Both operand fragments are contiguous 16B loads per lane per K-step.
// blockDim = 128 (4 wave32's). grid = (M/16, Ncols/64); wave w handles col tile 4*by+w.
// Caller guarantees M%16==0, K%32==0, Ncols%64==0 -> EXEC all ones, no masking.
template <typename OutT>
__global__ void wmma_gemm_kernel(const _Float16* __restrict__ A,
                                 const _Float16* __restrict__ Bt,
                                 const float*    __restrict__ bias,
                                 OutT*           __restrict__ C,
                                 int K, int Ncols) {
    const int lane = threadIdx.x & 31;
    const int wave = threadIdx.x >> 5;
    const int l15  = lane & 15;
    const int hsel = lane >> 4;           // 0: K 0..7 / 16..23 ; 1: K 8..15 / 24..31
    const int row   = blockIdx.x * 16 + l15;        // A row held by this lane
    const int tileN = blockIdx.y * 4 + wave;
    const int col   = tileN * 16 + l15;             // B/C/D column held by this lane
    const int kb    = hsel * 8;

    const _Float16* ap = A  + (size_t)row * K + kb;
    const _Float16* bp = Bt + (size_t)col * K + kb;

    v8f acc = {};
    for (int k0 = 0; k0 < K; k0 += 32) {
        v8h a_lo = *(const v8h*)(ap + k0);
        v8h a_hi = *(const v8h*)(ap + k0 + 16);
        v8h b_lo = *(const v8h*)(bp + k0);
        v8h b_hi = *(const v8h*)(bp + k0 + 16);
        v16h a, b;
#pragma unroll
        for (int i = 0; i < 8; ++i) {
            a[i] = a_lo[i]; a[8 + i] = a_hi[i];
            b[i] = b_lo[i]; b[8 + i] = b_hi[i];
        }
        acc = __builtin_amdgcn_wmma_f32_16x16x32_f16(false, a, false, b, (short)0, acc,
                                                     false, false);
    }
    const float bv = bias[col];
    OutT* cp = C + (size_t)(blockIdx.x * 16 + hsel * 8) * Ncols + col;
#pragma unroll
    for (int v = 0; v < 8; ++v) cp[(size_t)v * Ncols] = (OutT)(acc[v] + bv);
}

// ---------------------------------------------------------------- edge passes

// alpha[e,h] = (1/sqrt(dout)) * sum_c q[dst,h,c] * (k[src,h,c] + e[e,h,c]);  segment-max into amax
__global__ void edge_alpha_kernel(const int* __restrict__ ei,
                                  const _Float16* __restrict__ q,
                                  const _Float16* __restrict__ k,
                                  const float* __restrict__ ea,
                                  const float* __restrict__ we,
                                  float* __restrict__ alpha,
                                  float* __restrict__ amax,
                                  int dout, int hd, float scale) {
    int t = blockIdx.x * blockDim.x + threadIdx.x;
    if (t >= NEDGES * NHEADS) return;
    int e = t >> 2, h = t & 3;
    int src = ei[e], dst = ei[NEDGES + e];
    float eav[16];
#pragma unroll
    for (int j = 0; j < 16; ++j) eav[j] = ea[e * 16 + j];
    const _Float16* qp = q + (size_t)dst * hd + h * dout;
    const _Float16* kp = k + (size_t)src * hd + h * dout;
    const float*    wp = we + h * dout;
    float s = 0.0f;
    for (int c = 0; c < dout; ++c) {
        float ev = 0.0f;
#pragma unroll
        for (int j = 0; j < 16; ++j) ev += eav[j] * wp[(size_t)j * hd + c];
        s += (float)qp[c] * ((float)kp[c] + ev);
    }
    s *= scale;
    alpha[t] = s;
    atomicMaxF(&amax[dst * NHEADS + h], s);
}

// ex = exp(alpha - amax[dst]);  alpha <- ex;  den[dst] += ex
__global__ void edge_exp_kernel(const int* __restrict__ ei,
                                float* __restrict__ alpha,
                                const float* __restrict__ amax,
                                float* __restrict__ den) {
    int t = blockIdx.x * blockDim.x + threadIdx.x;
    if (t >= NEDGES * NHEADS) return;
    int e = t >> 2, h = t & 3;
    int dst = ei[NEDGES + e];
    float ex = __expf(alpha[t] - amax[dst * NHEADS + h]);
    alpha[t] = ex;
    atomicAdd(&den[dst * NHEADS + h], ex);
}

// agg[dst,h,c] += (v[src,h,c] + e[e,h,c]) * ex/den[dst,h]   (thread per (edge, channel))
__global__ void edge_agg_kernel(const int* __restrict__ ei,
                                const _Float16* __restrict__ v,
                                const float* __restrict__ ea,
                                const float* __restrict__ we,
                                const float* __restrict__ ex,
                                const float* __restrict__ den,
                                float* __restrict__ agg,
                                int dout, int hd) {
    int t = blockIdx.x * blockDim.x + threadIdx.x;
    if (t >= NEDGES * dout) return;
    int e = t / dout, c = t - e * dout;
    int src = ei[e], dst = ei[NEDGES + e];
    float eav[16];
#pragma unroll
    for (int j = 0; j < 16; ++j) eav[j] = ea[e * 16 + j];
#pragma unroll
    for (int h = 0; h < NHEADS; ++h) {
        float ev = 0.0f;
#pragma unroll
        for (int j = 0; j < 16; ++j) ev += eav[j] * we[(size_t)j * hd + h * dout + c];
        float attn = ex[e * NHEADS + h] / den[dst * NHEADS + h];
        float val  = ((float)v[(size_t)src * hd + h * dout + c] + ev) * attn;
        atomicAdd(&agg[(size_t)dst * hd + h * dout + c], val);
    }
}

// out = ELU(mean_h agg + skip)        (skip already contains x@ws + bs from GEMM epilogue)
__global__ void node_finalize_kernel(const float* __restrict__ agg,
                                     const float* __restrict__ skip,
                                     float* __restrict__ out,
                                     int dout, int hd) {
    int t = blockIdx.x * blockDim.x + threadIdx.x;
    if (t >= NNODES * dout) return;
    int n = t / dout, c = t - n * dout;
    float s = 0.0f;
#pragma unroll
    for (int h = 0; h < NHEADS; ++h) s += agg[(size_t)n * hd + h * dout + c];
    float val = s * 0.25f + skip[t];
    out[t] = (val > 0.0f) ? val : (__expf(val) - 1.0f);
}

// ---------------------------------------------------------------- pooling

__global__ void pool_accum_kernel(const float* __restrict__ h,
                                  const int* __restrict__ batch,
                                  float* __restrict__ sum,
                                  float* __restrict__ cnt) {
    int t = blockIdx.x * blockDim.x + threadIdx.x;
    if (t >= NNODES * 64) return;
    int n = t >> 6, c = t & 63;
    int g = batch[n];
    atomicAdd(&sum[g * 64 + c], h[t]);
    if (c == 0) atomicAdd(&cnt[g], 1.0f);
}

__global__ void pool_final_kernel(const float* __restrict__ sum,
                                  const float* __restrict__ cnt,
                                  float* __restrict__ out) {
    int t = blockIdx.x * blockDim.x + threadIdx.x;
    if (t >= NGRAPHS * 64) return;
    out[t] = sum[t] / fmaxf(cnt[t >> 6], 1.0f);
}

// ---------------------------------------------------------------- launcher

extern "C" void kernel_launch(void* const* d_in, const int* in_sizes, int n_in,
                              void* d_out, int out_size, void* d_ws, size_t ws_size,
                              hipStream_t stream) {
    const float* x     = (const float*)d_in[0];
    const int*   ei    = (const int*)  d_in[1];   // [2, E]: row0 src, row1 dst
    const float* ea    = (const float*)d_in[2];   // [E, 16]
    const int*   batch = (const int*)  d_in[3];   // [N]

    // bump allocator over workspace
    char*  ws  = (char*)d_ws;
    size_t off = 0;
    auto alloc = [&](size_t bytes) -> void* {
        void* p = ws + off;
        off += (bytes + 255) & ~(size_t)255;
        return p;
    };
    _Float16* hh   = (_Float16*)alloc((size_t)NNODES * 128 * sizeof(_Float16));
    _Float16* wqh  = (_Float16*)alloc((size_t)128 * 512 * sizeof(_Float16));
    _Float16* wkh  = (_Float16*)alloc((size_t)128 * 512 * sizeof(_Float16));
    _Float16* wvh  = (_Float16*)alloc((size_t)128 * 512 * sizeof(_Float16));
    _Float16* wsh  = (_Float16*)alloc((size_t)128 * 128 * sizeof(_Float16));
    _Float16* qb   = (_Float16*)alloc((size_t)NNODES * 512 * sizeof(_Float16));
    _Float16* kb   = (_Float16*)alloc((size_t)NNODES * 512 * sizeof(_Float16));
    _Float16* vb   = (_Float16*)alloc((size_t)NNODES * 512 * sizeof(_Float16));
    float* skip  = (float*)alloc((size_t)NNODES * 128 * sizeof(float));
    float* alpha = (float*)alloc((size_t)NEDGES * NHEADS * sizeof(float));
    float* amax  = (float*)alloc((size_t)NNODES * NHEADS * sizeof(float));
    float* den   = (float*)alloc((size_t)NNODES * NHEADS * sizeof(float));
    float* agg   = (float*)alloc((size_t)NNODES * 512 * sizeof(float));
    float* actA  = (float*)alloc((size_t)NNODES * 128 * sizeof(float));
    float* actB  = (float*)alloc((size_t)NNODES * 128 * sizeof(float));
    float* psum  = (float*)alloc((size_t)NGRAPHS * 64 * sizeof(float));
    float* pcnt  = (float*)alloc((size_t)NGRAPHS * sizeof(float));

    auto cvt = [&](const float* s, _Float16* d, int n) {
        cvt_f32_f16_kernel<<<(n + 255) / 256, 256, 0, stream>>>(s, d, n);
    };
    auto cvtT = [&](const float* w, _Float16* wt, int K, int Ncols) {
        int n = K * Ncols;
        cvt_transpose_f32_f16_kernel<<<(n + 255) / 256, 256, 0, stream>>>(w, wt, K, Ncols);
    };
    auto fill = [&](float* p, float v, int n) {
        fill_kernel<<<(n + 255) / 256, 256, 0, stream>>>(p, v, n);
    };

    const int dins[3]  = {128, 64, 128};
    const int douts[3] = {64, 128, 64};
    const float* hin = x;
    float* houts[3] = {actA, actB, actA};

    for (int l = 0; l < 3; ++l) {
        const int din = dins[l], dout = douts[l], hd = NHEADS * dout;
        const int base = 4 + l * 9;  // setup_inputs() dict order: wq wk wv we ws bq bk bv bs
        const float* wq  = (const float*)d_in[base + 0];
        const float* wk  = (const float*)d_in[base + 1];
        const float* wv  = (const float*)d_in[base + 2];
        const float* we  = (const float*)d_in[base + 3];
        const float* wsp = (const float*)d_in[base + 4];
        const float* bq  = (const float*)d_in[base + 5];
        const float* bk  = (const float*)d_in[base + 6];
        const float* bv  = (const float*)d_in[base + 7];
        const float* bs  = (const float*)d_in[base + 8];

        // activations f32 -> f16 (row-major A), weights f32 -> f16 transposed (col-major B)
        cvt(hin, hh, NNODES * din);
        cvtT(wq, wqh, din, hd);
        cvtT(wk, wkh, din, hd);
        cvtT(wv, wvh, din, hd);
        cvtT(wsp, wsh, din, dout);

        // node GEMMs via v_wmma_f32_16x16x32_f16
        dim3 blk(128);
        dim3 gq(NNODES / 16, hd / 64);
        wmma_gemm_kernel<_Float16><<<gq, blk, 0, stream>>>(hh, wqh, bq, qb, din, hd);
        wmma_gemm_kernel<_Float16><<<gq, blk, 0, stream>>>(hh, wkh, bk, kb, din, hd);
        wmma_gemm_kernel<_Float16><<<gq, blk, 0, stream>>>(hh, wvh, bv, vb, din, hd);
        dim3 gs(NNODES / 16, dout / 64);
        wmma_gemm_kernel<float><<<gs, blk, 0, stream>>>(hh, wsh, bs, skip, din, dout);

        // segment softmax + aggregation
        fill(amax, -INFINITY, NNODES * NHEADS);
        fill(den, 0.0f, NNODES * NHEADS);
        fill(agg, 0.0f, NNODES * hd);

        const float scale = 1.0f / sqrtf((float)dout);
        const int na = NEDGES * NHEADS;
        edge_alpha_kernel<<<(na + 255) / 256, 256, 0, stream>>>(ei, qb, kb, ea, we, alpha,
                                                                amax, dout, hd, scale);
        edge_exp_kernel<<<(na + 255) / 256, 256, 0, stream>>>(ei, alpha, amax, den);
        const int ng = NEDGES * dout;
        edge_agg_kernel<<<(ng + 255) / 256, 256, 0, stream>>>(ei, vb, ea, we, alpha, den,
                                                              agg, dout, hd);
        const int nf = NNODES * dout;
        node_finalize_kernel<<<(nf + 255) / 256, 256, 0, stream>>>(agg, skip, houts[l],
                                                                   dout, hd);
        hin = houts[l];
    }

    // global mean pool
    fill(psum, 0.0f, NGRAPHS * 64);
    fill(pcnt, 0.0f, NGRAPHS);
    pool_accum_kernel<<<(NNODES * 64 + 255) / 256, 256, 0, stream>>>(hin, batch, psum, pcnt);
    pool_final_kernel<<<(NGRAPHS * 64 + 255) / 256, 256, 0, stream>>>(psum, pcnt,
                                                                      (float*)d_out);
}
